// BasicBlock_41729902248084
// MI455X (gfx1250) — matrix operand order repaired and checked by reading, not compiled, not run on hardware
//
#include <hip/hip_runtime.h>

typedef __attribute__((ext_vector_type(8)))  float  v8f;
typedef __attribute__((ext_vector_type(16))) __bf16 v16bf;
typedef __attribute__((ext_vector_type(8)))  __bf16 v8bf;

#define BATCH 16
#define CIN   128
#define HH    56
#define WW    56
#define PH    58            // padded H (1px halo)
#define PW    58            // padded W
#define HWSZ  (HH*WW)       // 3136
#define MTOT  (BATCH*HWSZ)  // 50176
#define PELEM ((size_t)BATCH*PH*PW*CIN)   // padded activation elements (6,889,472)
#define NTAP  9
#define EPSV  1e-5f

// ---------------- helper kernels ----------------
__global__ __launch_bounds__(256) void k_zero(float* p, int n) {
  int i = blockIdx.x * 256 + threadIdx.x;
  if (i < n) p[i] = 0.f;
}

__global__ __launch_bounds__(256) void k_zero16(uint4* p, int n) {  // 16B-granular fill
  int i = blockIdx.x * 256 + threadIdx.x;
  if (i < n) p[i] = uint4{0u, 0u, 0u, 0u};
}

// per-tap psum-quant scales sq = wa/pa (hoists IEEE div out of conv loop)
__global__ void k_prep_scale(const float* wa1, const float* pa1,
                             const float* wa2, const float* pa2,
                             float* sq1, float* sq2) {
  int i = threadIdx.x;
  if (i < NTAP)        sq1[i] = wa1[i] / pa1[i];
  else if (i < 2*NTAP) sq2[i - NTAP] = wa2[i - NTAP] / pa2[i - NTAP];
}

// w_int[k,o,c] = (bf16) clamp(rint(w/alpha_k), -4, 3)   (exact small integers in bf16)
__global__ __launch_bounds__(256) void k_prep_w(const float* __restrict__ w,
                                                const float* __restrict__ wa,
                                                __bf16* __restrict__ out, int n) {
  int i = blockIdx.x * 256 + threadIdx.x;
  if (i >= n) return;
  int k = i / (CIN * CIN);
  float q = __builtin_rintf(w[i] / wa[k]);
  q = fminf(fmaxf(q, -4.f), 3.f);
  out[i] = (__bf16)q;
}

// x [B,C,H,W] f32 -> padded xb [B,PH,PW,C] bf16 interior; LDS-tiled transpose per (b,h)
__global__ __launch_bounds__(256) void k_prep_x(const float* __restrict__ x,
                                                __bf16* __restrict__ xb) {
  __shared__ __bf16 tile[WW * 129];
  int bh = blockIdx.x, b = bh / HH, h = bh % HH;
  const float* src = x + (size_t)b * CIN * HWSZ + h * WW;
  for (int i = threadIdx.x; i < WW * CIN; i += 256) {
    int c = i / WW, w = i % WW;                 // coalesced read along w
    tile[w * 129 + c] = (__bf16)src[(size_t)c * HWSZ + w];
  }
  __syncthreads();
  __bf16* dst = xb + ((size_t)(b * PH + h + 1) * PW + 1) * CIN;  // interior row
  for (int i = threadIdx.x; i < WW * CIN; i += 256) {
    int w = i / CIN, c = i % CIN;               // coalesced write along c
    dst[i] = tile[w * 129 + c];
  }
}

// ---------------- split-conv 3x3 as 9 shifted 1x1 bf16 WMMA GEMMs ----------------
// act : [B,PH,PW,128] zero-padded bf16 ; wint : [9,128,128] bf16 integer weights
// sqv : [9] = wa/pa ; pav : [9] ; out : [M,128] f32
// Weights staged per dj-group (3 taps, 96KB) in LDS in WMMA operand order:
//   wlds[t3][kc][j][lane][0..15]  (32B per lane slot -> conflict-free ds_load_b128)
__global__ __launch_bounds__(256)
void k_conv_split(const __bf16* __restrict__ act,
                  const __bf16* __restrict__ wint,
                  const float*  __restrict__ sqv,
                  const float*  __restrict__ pav,
                  float* __restrict__ outp)
{
  __shared__ __align__(32) __bf16 wlds[3 * 4 * 8 * 32 * 16];   // 96 KB
  const int tid  = threadIdx.x;
  const int lane = tid & 31;
  const int wid  = tid >> 5;            // 8 waves: 4 (M) x 2 (N)
  const int mw   = wid & 3;
  const int nw   = wid >> 2;
  const int m_base = blockIdx.x * 64 + mw * 16;
  const int lm    = lane & 15;
  const int khalf = lane >> 4;          // A: 0 -> K 0-7/16-23 ; 1 -> K 8-15/24-31
  const int m  = m_base + lm;
  const int b  = m / HWSZ;
  const int hw = m % HWSZ;
  const int h  = hw / WW;
  const int w  = hw % WW;

  // lane's halo-safe A base: padded coords (h, w) correspond to tap (di=0,dj=0)
  const __bf16* arow0 = act + ((size_t)(b * PH + h) * PW + w) * CIN;

  v8f run[4] = {};                      // fp32 running sums of quantized psums

  for (int dj = 0; dj < 3; ++dj) {      // w-shift (rolled; LDS-staged weights per group)
    __syncthreads();                    // previous group's LDS reads complete
    {
      const __bf16* wsrc = wint + (size_t)(dj * 3) * CIN * CIN;
      // 6144 slots x 16B; 24 slots per thread
      for (int s = tid; s < 3 * 4 * 8 * 32 * 2; s += 256) {
        const int halfb = s & 1;
        const int ln    = (s >> 1) & 31;
        const int j     = (s >> 6) & 7;
        const int kc    = (s >> 9) & 3;
        const int t3    = s >> 11;                 // 0..2
        const int lmm = ln & 15, kh = ln >> 4;
        const __bf16* src = wsrc + ((size_t)t3 * CIN + j * 16 + lmm) * CIN
                                 + kc * 32 + kh * 8 + halfb * 16;
        *(uint4*)(wlds + (size_t)s * 8) = *(const uint4*)src;
      }
    }
    __syncthreads();

#pragma unroll
    for (int di = 0; di < 3; ++di) {    // h-shift: compile-time immediate offsets
      const int tap = dj * 3 + di;      // reference tap order: dj outer, di inner
      const float sq = sqv[tap];        // uniform -> scalar loads
      const float pv = pav[tap];
      const __bf16* arow = arow0 + dj * CIN + di * (PW * CIN);

      v8f ps[4] = {};
#pragma unroll
      for (int kc = 0; kc < 4; ++kc) {             // K = 128 in chunks of 32
        const int ca = kc * 32 + khalf * 8;        // A-layout K split (interleaved 8s)
        v8bf alo = *(const v8bf*)(arow + ca);      // branchless: halo pre-zeroed
        v8bf ahi = *(const v8bf*)(arow + ca + 16);
        v16bf A = __builtin_shufflevector(alo, ahi,
                   0,1,2,3,4,5,6,7,8,9,10,11,12,13,14,15);
#pragma unroll
        for (int sub = 0; sub < 4; ++sub) {
          const int j = nw * 4 + sub;              // global N-subtile 0..7
          const __bf16* bl = wlds + ((size_t)(((di * 4 + kc) * 8 + j) * 32) + lane) * 16;
          v16bf Bm = *(const v16bf*)bl;            // 32B LDS read, conflict-free
          ps[sub] = __builtin_amdgcn_wmma_f32_16x16x32_bf16(
              false, A, false, Bm, (short)0, ps[sub], false, false);
        }
      }
      // per-tap psum LSQ: q = clamp(rint(ps_int * wa/pa), -128, 127); run += q*pa
#pragma unroll
      for (int sub = 0; sub < 4; ++sub)
#pragma unroll
        for (int r = 0; r < 8; ++r) {
          float q = __builtin_rintf(ps[sub][r] * sq);
          q = fminf(fmaxf(q, -128.f), 127.f);
          run[sub][r] = fmaf(q, pv, run[sub][r]);
        }
    }
  }

  // C/D layout: VGPR r, lanes<16 -> (M=r, N=lm); lanes>=16 -> (M=r+8, N=lm)
#pragma unroll
  for (int sub = 0; sub < 4; ++sub) {
    const int o = nw * 64 + sub * 16 + lm;
#pragma unroll
    for (int r = 0; r < 8; ++r) {
      const int mm = m_base + r + khalf * 8;
      outp[(size_t)mm * CIN + o] = run[sub][r];
    }
  }
}

// ---------------- batchnorm statistics / finalize / activation / epilogue ----------------
__global__ __launch_bounds__(256) void k_stats(const float* __restrict__ p,
                                               float* __restrict__ sum,
                                               float* __restrict__ sq) {
  __shared__ float ls[CIN], lq[CIN];
  const int t = threadIdx.x;
  if (t < CIN) { ls[t] = 0.f; lq[t] = 0.f; }
  __syncthreads();
  const int rows = MTOT / 256;                 // 196 rows per block
  const int m0 = blockIdx.x * rows;
  const int o = t & 127, half = t >> 7;
  float s = 0.f, q = 0.f;
  for (int r = half; r < rows; r += 2) {
    float v = p[(size_t)(m0 + r) * CIN + o];   // coalesced
    s += v; q += v * v;
  }
  atomicAdd(&ls[o], s); atomicAdd(&lq[o], q);
  __syncthreads();
  if (t < CIN) { atomicAdd(&sum[t], ls[t]); atomicAdd(&sq[t], lq[t]); }
}

__global__ void k_bnfin(const float* sum, const float* sq, const float* g, const float* bb,
                        float* scale, float* bias) {
  int o = threadIdx.x;
  float m = sum[o] * (1.f / (float)MTOT);
  float v = sq[o] * (1.f / (float)MTOT) - m * m;
  float sc = g[o] / sqrtf(v + EPSV);
  scale[o] = sc;
  bias[o]  = bb[o] - m * sc;
}

// relu(bn1(ps)) -> bf16 into PADDED layout for conv2's halo-safe loads
__global__ __launch_bounds__(256) void k_act2(const float* __restrict__ ps,
                                              const float* __restrict__ scale,
                                              const float* __restrict__ bias,
                                              __bf16* __restrict__ y) {
  size_t i = (size_t)blockIdx.x * 256 + threadIdx.x;
  int o = (int)(i & 127);
  int mp = (int)(i >> 7);
  int b = mp / HWSZ, r = mp % HWSZ, h = r / WW, w = r % WW;
  float v = fmaf(ps[i], scale[o], bias[o]);
  y[((size_t)(b * PH + h + 1) * PW + (w + 1)) * CIN + o] = (__bf16)fmaxf(v, 0.f);
}

// out[b,o,h,w] = relu(bn2(ps[m,o]) + x[b,o,h,w]) ; LDS-tiled [M,O]->[B,O,H,W] transpose
__global__ __launch_bounds__(256) void k_final(const float* __restrict__ ps,
                                               const float* __restrict__ scale,
                                               const float* __restrict__ bias,
                                               const float* __restrict__ x,
                                               float* __restrict__ out) {
  __shared__ float tile[WW * 129];
  int bh = blockIdx.x, b = bh / HH, h = bh % HH;
  const float* src = ps + (size_t)bh * WW * CIN;
  for (int i = threadIdx.x; i < WW * CIN; i += 256) {
    int w = i / CIN, c = i % CIN;               // coalesced read along c
    tile[w * 129 + c] = src[i];
  }
  __syncthreads();
  for (int i = threadIdx.x; i < WW * CIN; i += 256) {
    int o = i / WW, w = i % WW;                 // coalesced write along w
    float v = fmaf(tile[w * 129 + o], scale[o], bias[o]);
    size_t gi = ((size_t)(b * CIN + o) * HH + h) * WW + w;
    out[gi] = fmaxf(v + x[gi], 0.f);
  }
}

// ---------------- launch ----------------
extern "C" void kernel_launch(void* const* d_in, const int* in_sizes, int n_in,
                              void* d_out, int out_size, void* d_ws, size_t ws_size,
                              hipStream_t stream) {
  const float* x   = (const float*)d_in[0];
  const float* w1  = (const float*)d_in[1];
  const float* wa1 = (const float*)d_in[2];
  const float* pa1 = (const float*)d_in[3];
  const float* g1  = (const float*)d_in[4];
  const float* b1  = (const float*)d_in[5];
  const float* w2  = (const float*)d_in[6];
  const float* wa2 = (const float*)d_in[7];
  const float* pa2 = (const float*)d_in[8];
  const float* g2  = (const float*)d_in[9];
  const float* b2  = (const float*)d_in[10];
  float* out = (float*)d_out;

  char* ws = (char*)d_ws;
  float* stats  = (float*)ws;                              // 2048 floats reserved
  float* sum1 = stats,        *sq1s = stats + 128;
  float* sum2 = stats + 256,  *sq2s = stats + 384;
  float* scale1 = stats + 512, *bias1 = stats + 640;
  float* scale2 = stats + 768, *bias2 = stats + 896;
  float* sqw1 = stats + 1024,  *sqw2 = stats + 1040;       // per-tap wa/pa scales
  const size_t wbytes = (size_t)NTAP * CIN * CIN * 2;      // 294912
  const size_t pbytes = PELEM * 2;                         // 13,778,944
  __bf16* wint1 = (__bf16*)(ws + 8192);
  __bf16* wint2 = (__bf16*)(ws + 8192 + wbytes);
  __bf16* xbp   = (__bf16*)(ws + 8192 + 2 * wbytes);       // padded bf16 act, conv1 in
  __bf16* y1bp  = (__bf16*)(ws + 8192 + 2 * wbytes + pbytes);   // padded bf16, conv2 in
  float*  psb   = (float*)(ws + 8192 + 2 * wbytes + 2 * pbytes); // [M,128] f32 (reused)

  const int wn = NTAP * CIN * CIN;
  const int pz = (int)(pbytes / 16);                       // 861,184 uint4 per buffer
  k_zero<<<2, 256, 0, stream>>>(stats, 512);
  k_prep_scale<<<1, 32, 0, stream>>>(wa1, pa1, wa2, pa2, sqw1, sqw2);
  k_zero16<<<(pz + 255) / 256, 256, 0, stream>>>((uint4*)xbp, pz);
  k_zero16<<<(pz + 255) / 256, 256, 0, stream>>>((uint4*)y1bp, pz);
  k_prep_w<<<(wn + 255) / 256, 256, 0, stream>>>(w1, wa1, wint1, wn);
  k_prep_w<<<(wn + 255) / 256, 256, 0, stream>>>(w2, wa2, wint2, wn);
  k_prep_x<<<BATCH * HH, 256, 0, stream>>>(x, xbp);

  k_conv_split<<<MTOT / 64, 256, 0, stream>>>(xbp, wint1, sqw1, pa1, psb);
  k_stats<<<256, 256, 0, stream>>>(psb, sum1, sq1s);
  k_bnfin<<<1, 128, 0, stream>>>(sum1, sq1s, g1, b1, scale1, bias1);
  k_act2<<<(MTOT * CIN) / 256, 256, 0, stream>>>(psb, scale1, bias1, y1bp);

  k_conv_split<<<MTOT / 64, 256, 0, stream>>>(y1bp, wint2, sqw2, pa2, psb);
  k_stats<<<256, 256, 0, stream>>>(psb, sum2, sq2s);
  k_bnfin<<<1, 128, 0, stream>>>(sum2, sq2s, g2, b2, scale2, bias2);
  k_final<<<BATCH * HH, 256, 0, stream>>>(psb, scale2, bias2, x, out);
}
